// WindowMSA_77060303225102
// MI455X (gfx1250) — compile-verified, matrix operand order checked by reading, not run on hardware
//
#include <hip/hip_runtime.h>
#include <hip/hip_bf16.h>

typedef __attribute__((ext_vector_type(16))) __bf16 v16bf;
typedef __attribute__((ext_vector_type(8)))  float  v8f;

#define NTOK   49
#define EMBED  256
#define HEADS  8
#define NPAD   64
#define NWMASK 16

// LDS strides (bf16 elements), padded to avoid bank conflicts
#define SQ   258   // q_s / v_s (also y)
#define SK   258   // k_s (inside union)
#define SXB  260   // x tile (inside union)
#define SSLAB 65   // per-wave attn-probability slab
#define UMEM_ELEMS (8 * NPAD * SSLAB)   // 33280 elems = 66560 B (covers k_s + xb)

// workspace layout (bytes)
#define WS_QKVW      0
#define WS_QKVW_SZ   (8 * 48 * 32 * 16 * 2)   // 393216: 8 ksteps x 48 ntiles, frag-swizzled bf16
#define WS_PROJW     (WS_QKVW + WS_QKVW_SZ)
#define WS_PROJW_SZ  (8 * 16 * 32 * 16 * 2)   // 131072
#define WS_BIAS      (WS_PROJW + WS_PROJW_SZ) // 8*49*49 floats

static __device__ __forceinline__ int kk_of(int v, int half) {
  return (v < 4) ? (2 * v + 8 * half) : (16 + 2 * (v - 4) + 8 * half);
}

static __device__ __forceinline__ v8f wmma_bf16(v16bf a, v16bf b, v8f c) {
  return __builtin_amdgcn_wmma_f32_16x16x32_bf16(false, a, false, b, (short)0, c, false, false);
}

// A fragment (16x32 bf16): rows m0..m0+15, K cols k0..k0+31, LDS row stride S
static __device__ __forceinline__ v16bf load_a_lds(const __bf16* base, int S, int m0, int k0) {
  const int lane = threadIdx.x & 31;
  const int half = lane >> 4, r = lane & 15;
  const __bf16* row = base + (m0 + r) * S + k0;
  v16bf a;
#pragma unroll
  for (int v = 0; v < 8; ++v) {
    const int kk = kk_of(v, half);
    a[2 * v]     = row[kk];
    a[2 * v + 1] = row[kk + 1];
  }
  return a;
}

// B fragment (32x16) where B[k][n] = src[(n0+n)*S + c0 + k]  (i.e. src holds B transposed: k-matrix)
static __device__ __forceinline__ v16bf load_bt_lds(const __bf16* base, int S, int n0, int c0) {
  const int lane = threadIdx.x & 31;
  const int half = lane >> 4, n = lane & 15;
  const __bf16* row = base + (n0 + n) * S + c0;
  v16bf b;
#pragma unroll
  for (int v = 0; v < 8; ++v) {
    const int kk = kk_of(v, half);
    b[2 * v]     = row[kk];
    b[2 * v + 1] = row[kk + 1];
  }
  return b;
}

// B fragment (32x16) where B[k][n] = src[(k0+k)*S + n0 + n]  (row-major K x N: v-matrix)
static __device__ __forceinline__ v16bf load_b_rm_lds(const __bf16* base, int S, int k0, int n0) {
  const int lane = threadIdx.x & 31;
  const int half = lane >> 4, n = lane & 15;
  v16bf b;
#pragma unroll
  for (int v = 0; v < 8; ++v) {
    const int kk = kk_of(v, half);
    b[2 * v]     = base[(k0 + kk)     * S + n0 + n];
    b[2 * v + 1] = base[(k0 + kk + 1) * S + n0 + n];
  }
  return b;
}

// B fragment from pre-swizzled global weights: one contiguous 32B load per lane
static __device__ __forceinline__ v16bf load_b_frag_global(const __bf16* wf, int f) {
  const int lane = threadIdx.x & 31;
  return *(const v16bf*)(wf + ((size_t)(f * 32 + lane)) * 16);
}

// ---------------- prep kernels ----------------

// Swizzle fp32 weights [K=256][Ncols] into bf16 B-fragment layout.
__global__ void prep_wfrag(const float* __restrict__ w, __bf16* __restrict__ frag,
                           int Ncols, int NT, int total) {
  int t = blockIdx.x * 256 + threadIdx.x;
  if (t >= total) return;
  const int e = t & 15;
  const int lane = (t >> 4) & 31;
  const int f = t >> 9;
  const int nt = f % NT, ks = f / NT;
  const int v = e >> 1, p = e & 1, hf = lane >> 4, n = lane & 15;
  const int kk = ((v < 4) ? (2 * v + 8 * hf) : (16 + 2 * (v - 4) + 8 * hf)) + p;
  const int K = ks * 32 + kk;
  const int N = nt * 16 + n;
  frag[t] = (__bf16)w[(size_t)K * Ncols + N];
}

// bias[h][n][m] = bias_table[rel_index[n*49+m]][h]
__global__ void prep_bias(const float* __restrict__ table, const int* __restrict__ relidx,
                          float* __restrict__ bias) {
  int t = blockIdx.x * 256 + threadIdx.x;
  if (t >= HEADS * NTOK * NTOK) return;
  const int h = t / (NTOK * NTOK);
  const int r = t - h * (NTOK * NTOK);
  bias[t] = table[relidx[r] * HEADS + h];
}

// ---------------- fused window-MSA kernel ----------------

__global__ __launch_bounds__(256)
void winmsa_kernel(const float* __restrict__ x, const float* __restrict__ mask,
                   const float* __restrict__ qkv_b, const float* __restrict__ proj_b,
                   const __bf16* __restrict__ wq, const __bf16* __restrict__ wp,
                   const float* __restrict__ bias_hnn, float* __restrict__ out) {
  __shared__ __bf16 q_s[NPAD * SQ];     // q (scaled); later aliased as y = attn*v
  __shared__ __bf16 v_s[NPAD * SQ];
  __shared__ __bf16 u_mem[UMEM_ELEMS];  // union { k_s + x tile | 8 attn slabs }
  __bf16* k_s = u_mem;
  __bf16* xb  = u_mem + NPAD * SK;

  const int b    = blockIdx.x;
  const int tid  = threadIdx.x;
  const int wave = tid >> 5;
  const int lane = tid & 31;
  const int half = lane >> 4;
  const int ln   = lane & 15;

  // ---- Phase 1: x -> bf16 LDS tile, zero-padded to 64 rows ----
  const float* xp = x + (size_t)b * NTOK * EMBED;
  for (int i = tid; i < NPAD * EMBED; i += 256) {
    const int t = i >> 8;       // token
    const int c = i & 255;      // channel
    const float vv = (t < NTOK) ? xp[t * EMBED + c] : 0.0f;
    xb[t * SXB + c] = (__bf16)vv;
  }
  __syncthreads();

  // ---- Phase 2: QKV GEMM (each wave: 6 n-tiles of 16 over 768 cols) ----
  const float scale = 0.17677669529663687f;  // hd^-0.5
  for (int mt = 0; mt < 4; ++mt) {
    v8f acc[6];
#pragma unroll
    for (int j = 0; j < 6; ++j) acc[j] = (v8f)0.0f;
    for (int ks = 0; ks < 8; ++ks) {
      const v16bf a = load_a_lds(xb, SXB, mt * 16, ks * 32);
#pragma unroll
      for (int j = 0; j < 6; ++j) {
        const int nt = wave * 6 + j;
        const v16bf bf = load_b_frag_global(wq, ks * 48 + nt);
        acc[j] = wmma_bf16(a, bf, acc[j]);
      }
    }
#pragma unroll
    for (int j = 0; j < 6; ++j) {
      const int col = (wave * 6 + j) * 16 + ln;
      const float bias = qkv_b[col];
#pragma unroll
      for (int v = 0; v < 8; ++v) {
        const int r = mt * 16 + v + 8 * half;
        const float val = acc[j][v] + bias;
        if (col < 256)       q_s[r * SQ + col]         = (__bf16)(val * scale);
        else if (col < 512)  k_s[r * SK + (col - 256)] = (__bf16)val;
        else                 v_s[r * SQ + (col - 512)] = (__bf16)val;
      }
    }
  }
  __syncthreads();

  // ---- Phase 3: scores for head h = wave (q*scale @ k^T), all tiles in regs ----
  const int h = wave;
  const int w = b & (NWMASK - 1);
  const float* mk = mask + (size_t)w * NTOK * NTOK;
  const float* bs = bias_hnn + (size_t)h * NTOK * NTOK;

  v8f sc[4][4];
  for (int mt = 0; mt < 4; ++mt) {
    const v16bf qa = load_a_lds(q_s, SQ, mt * 16, h * 32);
#pragma unroll
    for (int nt = 0; nt < 4; ++nt) {
      v8f z = (v8f)0.0f;
      const v16bf kb = load_bt_lds(k_s, SK, nt * 16, h * 32);
      sc[mt][nt] = wmma_bf16(qa, kb, z);
    }
  }
  __syncthreads();  // all waves done with q_s/k_s/xb -> slabs may clobber union

  __bf16* slab = u_mem + wave * (NPAD * SSLAB);
  const float NEG_INF = -__builtin_inff();
  const float LOG2E_UNUSED = 0.0f; (void)LOG2E_UNUSED;

  for (int mt = 0; mt < 4; ++mt) {
    // bias + attention mask; kill padded columns
#pragma unroll
    for (int nt = 0; nt < 4; ++nt) {
#pragma unroll
      for (int v = 0; v < 8; ++v) {
        const int n = mt * 16 + v + 8 * half;
        const int m = nt * 16 + ln;
        float t = sc[mt][nt][v];
        if (n < NTOK && m < NTOK) t += bs[n * NTOK + m] + mk[n * NTOK + m];
        if (m >= NTOK) t = NEG_INF;
        sc[mt][nt][v] = t;
      }
    }
    // row max (reduce over 4 n-tiles then over 16 column-lanes)
    float rmax[8], rsum[8];
#pragma unroll
    for (int v = 0; v < 8; ++v) {
      float m0 = fmaxf(fmaxf(sc[mt][0][v], sc[mt][1][v]), fmaxf(sc[mt][2][v], sc[mt][3][v]));
      m0 = fmaxf(m0, __shfl_xor(m0, 1, 32));
      m0 = fmaxf(m0, __shfl_xor(m0, 2, 32));
      m0 = fmaxf(m0, __shfl_xor(m0, 4, 32));
      m0 = fmaxf(m0, __shfl_xor(m0, 8, 32));
      rmax[v] = m0;
    }
#pragma unroll
    for (int v = 0; v < 8; ++v) rsum[v] = 0.0f;
#pragma unroll
    for (int nt = 0; nt < 4; ++nt) {
#pragma unroll
      for (int v = 0; v < 8; ++v) {
        const float e = __expf(sc[mt][nt][v] - rmax[v]);
        sc[mt][nt][v] = e;
        rsum[v] += e;
      }
    }
#pragma unroll
    for (int v = 0; v < 8; ++v) {
      float s = rsum[v];
      s += __shfl_xor(s, 1, 32);
      s += __shfl_xor(s, 2, 32);
      s += __shfl_xor(s, 4, 32);
      s += __shfl_xor(s, 8, 32);
      rsum[v] = __frcp_rn(s);
    }
    // write bf16 probabilities to this wave's slab (padded cols become exact 0)
#pragma unroll
    for (int nt = 0; nt < 4; ++nt) {
#pragma unroll
      for (int v = 0; v < 8; ++v) {
        const int n = mt * 16 + v + 8 * half;
        const int m = nt * 16 + ln;
        slab[n * SSLAB + m] = (__bf16)(sc[mt][nt][v] * rsum[v]);
      }
    }
  }

  // ---- Phase 4: out_h = attn @ v_h ; y overwrites q_s (dead) ----
  for (int mt = 0; mt < 4; ++mt) {
    v8f oacc[2];
    oacc[0] = (v8f)0.0f;
    oacc[1] = (v8f)0.0f;
    for (int ks = 0; ks < 2; ++ks) {
      const v16bf aa = load_a_lds(slab, SSLAB, mt * 16, ks * 32);
#pragma unroll
      for (int dt = 0; dt < 2; ++dt) {
        const v16bf vb = load_b_rm_lds(v_s, SQ, ks * 32, h * 32 + dt * 16);
        oacc[dt] = wmma_bf16(aa, vb, oacc[dt]);
      }
    }
#pragma unroll
    for (int dt = 0; dt < 2; ++dt) {
#pragma unroll
      for (int v = 0; v < 8; ++v) {
        const int r = mt * 16 + v + 8 * half;
        q_s[r * SQ + h * 32 + dt * 16 + ln] = (__bf16)oacc[dt][v];
      }
    }
  }
  __syncthreads();  // all heads' y written

  // ---- Phase 5: out = y @ proj_w + proj_b (each wave: 2 n-tiles over 256 cols) ----
  for (int mt = 0; mt < 4; ++mt) {
    v8f pacc[2];
    pacc[0] = (v8f)0.0f;
    pacc[1] = (v8f)0.0f;
    for (int ks = 0; ks < 8; ++ks) {
      const v16bf aa = load_a_lds(q_s, SQ, mt * 16, ks * 32);
#pragma unroll
      for (int j = 0; j < 2; ++j) {
        const int nt = wave * 2 + j;
        const v16bf bf = load_b_frag_global(wp, ks * 16 + nt);
        pacc[j] = wmma_bf16(aa, bf, pacc[j]);
      }
    }
#pragma unroll
    for (int j = 0; j < 2; ++j) {
      const int col = (wave * 2 + j) * 16 + ln;
      const float pb = proj_b[col];
#pragma unroll
      for (int v = 0; v < 8; ++v) {
        const int r = mt * 16 + v + 8 * half;
        if (r < NTOK) out[((size_t)b * NTOK + r) * EMBED + col] = pacc[j][v] + pb;
      }
    }
  }
}

extern "C" void kernel_launch(void* const* d_in, const int* in_sizes, int n_in,
                              void* d_out, int out_size, void* d_ws, size_t ws_size,
                              hipStream_t stream) {
  (void)in_sizes; (void)n_in; (void)out_size; (void)ws_size;
  const float* x          = (const float*)d_in[0];
  const float* mask       = (const float*)d_in[1];
  const float* qkv_w      = (const float*)d_in[2];
  const float* qkv_b      = (const float*)d_in[3];
  const float* proj_w     = (const float*)d_in[4];
  const float* proj_b     = (const float*)d_in[5];
  const float* bias_table = (const float*)d_in[6];
  const int*   rel_index  = (const int*)d_in[7];
  float* out = (float*)d_out;

  unsigned char* ws = (unsigned char*)d_ws;
  __bf16* wq = (__bf16*)(ws + WS_QKVW);
  __bf16* wp = (__bf16*)(ws + WS_PROJW);
  float* bias_hnn = (float*)(ws + WS_BIAS);

  // weight swizzle + bias materialization (tiny; weights then live in L2)
  {
    const int total_q = 8 * 48 * 32 * 16;  // 196608
    prep_wfrag<<<(total_q + 255) / 256, 256, 0, stream>>>(qkv_w, wq, 768, 48, total_q);
    const int total_p = 8 * 16 * 32 * 16;  // 65536
    prep_wfrag<<<(total_p + 255) / 256, 256, 0, stream>>>(proj_w, wp, 256, 16, total_p);
    const int total_b = HEADS * NTOK * NTOK;
    prep_bias<<<(total_b + 255) / 256, 256, 0, stream>>>(bias_table, rel_index, bias_hnn);
  }

  const int B = 4096;
  winmsa_kernel<<<B, 256, 0, stream>>>(x, mask, qkv_b, proj_b, wq, wp, bias_hnn, out);
}
